// TransformerBlock_27668179320919
// MI455X (gfx1250) — compile-verified, matrix operand order checked by reading, not run on hardware
//
#include <hip/hip_runtime.h>
#include <hip/hip_bf16.h>

// ---------------------------------------------------------------------------
// Problem constants (from reference)
// ---------------------------------------------------------------------------
#define Bsz      4
#define Sq       2048
#define DIM      1024
#define HEADS    16
#define KVHEADS  4
#define HDIM     64
#define HIDDEN   2048
#define NEXP     8
#define TOPK     2
#define Ttok     (Bsz * Sq)          // 8192 tokens
#define NSLOT    (Ttok * TOPK)       // 16384 routed slots
#define EPSF     1e-5f
#define NEGINF   -1e9f

typedef __attribute__((ext_vector_type(16))) __bf16 v16bf;
typedef __attribute__((ext_vector_type(8)))  float  v8f;

// ---------------------------------------------------------------------------
// WMMA helpers (CDNA5 layouts, CDNA5 ISA §7.12.2)
// A 16x32 bf16 fragment: lane L holds row M=L&15; pair i (VGPR i) holds
//   K = (i<4 ? 2i : 8+2i) + (L>=16 ? 8 : 0)  and K+1.
// B 32x16 bf16 fragment: mirrored with N=L&15; LDS tile stored [N][K].
// C/D 16x16 f32: element v -> row = v + (L>=16 ? 8:0), col = L&15.
// ---------------------------------------------------------------------------
__device__ __forceinline__ v16bf load_frag(const __bf16* base, int ld, int r0,
                                           int k0, int lane) {
  int r  = r0 + (lane & 15);
  int kb = k0 + ((lane & 16) ? 8 : 0);
  const unsigned* p = (const unsigned*)(base + (size_t)r * ld + kb);
  union { v16bf v; unsigned u[8]; } f;
#pragma unroll
  for (int i = 0; i < 4; ++i) { f.u[i] = p[i]; f.u[i + 4] = p[i + 8]; }
  return f.v;
}

__device__ __forceinline__ v8f wmma_bf16(v16bf a, v16bf b, v8f c) {
  return __builtin_amdgcn_wmma_f32_16x16x32_bf16(false, a, false, b,
                                                 (short)0, c, false, false);
}

// ---------------------------------------------------------------------------
// Generic WMMA GEMM:  C[crow][0..N) = A[arow][0..Kd) x W
//   W layout: BKM==false -> W[N][Kd] (row-major (out,in), i.e. x @ W^T)
//             BKM==true  -> W[Kd][N] (K-major, used for w2)
//   expBase/expCount: per-expert slot range (blockIdx.z = expert); null -> 0/M.
//   rowMap: A-row gather (token list); null -> arow = base+m.
//   EPI: 0 none | 1 +res[crow][col] | 2 silu(acc)*C (in place) | 3 acc*sw[crow]
// Block: 256 threads = 8 waves (4 row x 2 col); wave tile 32x64;
// block tile 128x128; K-step 32; next K-tile staged in registers while the
// current one is computed from LDS (global latency hidden behind WMMA issue).
// ---------------------------------------------------------------------------
template <int EPI, bool BKM>
__global__ __launch_bounds__(256) void gemm_wmma_k(
    const float* __restrict__ A, const float* __restrict__ W,
    float* __restrict__ C, int M, int N, int Kd, size_t wstride,
    const int* __restrict__ expBase, const int* __restrict__ expCount,
    const int* __restrict__ rowMap, const float* __restrict__ res,
    const float* __restrict__ sw) {
  constexpr int BM = 128, BN = 128, BK = 32, LDT = BK + 2;
  __shared__ __bf16 As[BM][LDT];
  __shared__ __bf16 Bs[BN][LDT];

  const int e    = blockIdx.z;
  const int base = expBase ? expBase[e] : 0;
  const int cnt  = expCount ? expCount[e] : M;
  const int m0   = blockIdx.y * BM;
  if (m0 >= cnt) return;
  const float* Wp = W + (size_t)e * wstride;
  const int n0  = blockIdx.x * BN;
  const int tid = threadIdx.x;
  const int lane = tid & 31;
  const int wave = tid >> 5;
  const int wr = (wave >> 1) * 32;   // wave row offset (0/32/64/96)
  const int wc = (wave & 1) * 64;    // wave col offset (0/64)
  const int ln15 = lane & 15;
  const int hib  = (lane & 16) ? 8 : 0;

  v8f acc[2][4] = {};
  float aReg[16], bReg[16];

  // staged global loads of one 128x32 K-tile (A) and 128x32 (B)
  auto loadA = [&](int k0) {
#pragma unroll
    for (int i = 0; i < 16; ++i) {
      int el = tid + i * 256;
      int r = el >> 5, c = el & 31;
      float v = 0.0f;
      int m = m0 + r;
      if (m < cnt) {
        int ar = rowMap ? rowMap[base + m] : (base + m);
        v = A[(size_t)ar * Kd + k0 + c];
      }
      aReg[i] = v;
    }
  };
  auto loadB = [&](int k0) {
    if (!BKM) {
#pragma unroll
      for (int i = 0; i < 16; ++i) {
        int el = tid + i * 256;
        int r = el >> 5, c = el & 31;
        bReg[i] = Wp[(size_t)(n0 + r) * Kd + k0 + c];
      }
    } else {
#pragma unroll
      for (int i = 0; i < 16; ++i) {
        int el = tid + i * 256;
        int nn = el & 127, kk = el >> 7;
        bReg[i] = Wp[(size_t)(k0 + kk) * N + n0 + nn];
      }
    }
  };

  loadA(0);
  loadB(0);

  for (int k0 = 0; k0 < Kd; k0 += BK) {
    // commit staged tile to LDS (fp32 -> bf16)
#pragma unroll
    for (int i = 0; i < 16; ++i) {
      int el = tid + i * 256;
      As[el >> 5][el & 31] = (__bf16)aReg[i];
    }
    if (!BKM) {
#pragma unroll
      for (int i = 0; i < 16; ++i) {
        int el = tid + i * 256;
        Bs[el >> 5][el & 31] = (__bf16)bReg[i];
      }
    } else {
#pragma unroll
      for (int i = 0; i < 16; ++i) {
        int el = tid + i * 256;
        Bs[el & 127][el >> 7] = (__bf16)bReg[i];
      }
    }
    __syncthreads();

    // issue next K-tile global loads; they complete while WMMAs run
    if (k0 + BK < Kd) {
      loadA(k0 + BK);
      loadB(k0 + BK);
    }

    // 8 WMMAs per wave per K-step (2 A-frags x 4 B-frags)
    v16bf a0 = load_frag(&As[0][0], LDT, wr, 0, lane);
    v16bf a1 = load_frag(&As[0][0], LDT, wr + 16, 0, lane);
#pragma unroll
    for (int ni = 0; ni < 4; ++ni) {
      v16bf bb = load_frag(&Bs[0][0], LDT, wc + ni * 16, 0, lane);
      acc[0][ni] = wmma_bf16(a0, bb, acc[0][ni]);
      acc[1][ni] = wmma_bf16(a1, bb, acc[1][ni]);
    }
    __syncthreads();
  }

  // --- epilogue + store
#pragma unroll
  for (int mi = 0; mi < 2; ++mi)
#pragma unroll
    for (int ni = 0; ni < 4; ++ni) {
      const float* av = (const float*)&acc[mi][ni];
      int col = n0 + wc + ni * 16 + ln15;
#pragma unroll
      for (int v = 0; v < 8; ++v) {
        int m = m0 + wr + mi * 16 + v + hib;
        if (m < cnt) {
          size_t crow = (size_t)(base + m);
          float val = av[v];
          if (EPI == 1) val += res[crow * N + col];
          if (EPI == 2) {
            float u = C[crow * (size_t)N + col];
            val = (val / (1.0f + __expf(-val))) * u;   // silu(g) * u
          }
          if (EPI == 3) val *= sw[crow];
          C[crow * (size_t)N + col] = val;
        }
      }
    }
}

// ---------------------------------------------------------------------------
// RMSNorm: one block per token row of DIM elems
// ---------------------------------------------------------------------------
__global__ __launch_bounds__(256) void rmsnorm_k(const float* __restrict__ x,
                                                 const float* __restrict__ w,
                                                 float* __restrict__ y) {
  __shared__ float red[256];
  const size_t t = blockIdx.x;
  const int tid = threadIdx.x;
  float ss = 0.0f;
  for (int j = tid; j < DIM; j += 256) {
    float v = x[t * DIM + j];
    ss += v * v;
  }
  red[tid] = ss;
  __syncthreads();
  for (int off = 128; off > 0; off >>= 1) {
    if (tid < off) red[tid] += red[tid + off];
    __syncthreads();
  }
  float scale = rsqrtf(red[0] / (float)DIM + EPSF);
  for (int j = tid; j < DIM; j += 256)
    y[t * DIM + j] = x[t * DIM + j] * scale * w[j];
}

// ---------------------------------------------------------------------------
// RoPE (interleaved pairs), in place on [T][nh*64]
// ---------------------------------------------------------------------------
__global__ __launch_bounds__(256) void rope_k(float* __restrict__ buf, int nh,
                                              const float* __restrict__ cosb,
                                              const float* __restrict__ sinb) {
  int idx = blockIdx.x * 256 + threadIdx.x;
  int total = Ttok * nh * (HDIM / 2);
  if (idx >= total) return;
  int i = idx & 31;
  int h = (idx >> 5) % nh;
  int t = idx / (32 * nh);
  int s = t & (Sq - 1);
  float c = cosb[s * 32 + i], sn = sinb[s * 32 + i];
  float* p = buf + (size_t)t * nh * HDIM + h * HDIM + 2 * i;
  float x0 = p[0], x1 = p[1];
  p[0] = x0 * c - x1 * sn;
  p[1] = x0 * sn + x1 * c;
}

// ---------------------------------------------------------------------------
// Flash attention with WMMA. Grid (S/64, HEADS, B), 128 threads = 4 waves.
// Each wave owns 16 query rows; scores scaled by *8 (reference divides by
// HDIM**-0.5) and causal-masked with -1e9 before softmax.
// ---------------------------------------------------------------------------
__global__ __launch_bounds__(128) void attn_k(const float* __restrict__ q,
                                              const float* __restrict__ kbuf,
                                              const float* __restrict__ vbuf,
                                              float* __restrict__ att) {
  constexpr int LD = HDIM + 2;
  __shared__ __bf16 Qs[64][LD];
  __shared__ __bf16 Ks[64][LD];
  __shared__ __bf16 Vs[64][LD];   // Vs[dim][key]
  __shared__ __bf16 Ps[64][LD];   // Ps[row][key]

  const int b = blockIdx.z, h = blockIdx.y, q0 = blockIdx.x * 64;
  const int hk = h / (HEADS / KVHEADS);
  const int tid = threadIdx.x;
  const int lane = tid & 31, wave = tid >> 5;
  const int qr0 = wave * 16;
  const int ln15 = lane & 15, hib = (lane & 16) ? 8 : 0;

  // load Q tile
#pragma unroll
  for (int i = 0; i < 32; ++i) {
    int el = tid + i * 128;
    int r = el >> 6, d = el & 63;
    Qs[r][d] = (__bf16)q[(size_t)(b * Sq + q0 + r) * (HEADS * HDIM) +
                         h * HDIM + d];
  }
  __syncthreads();

  v8f o[4] = {};
  float mr[8], lr[8];
#pragma unroll
  for (int v = 0; v < 8; ++v) { mr[v] = -3e38f; lr[v] = 0.0f; }

  for (int kt0 = 0; kt0 <= q0; kt0 += 64) {
    // load K,V tiles (V transposed to [dim][key])
#pragma unroll
    for (int i = 0; i < 32; ++i) {
      int el = tid + i * 128;
      int key = el >> 6, d = el & 63;
      size_t g = (size_t)(b * Sq + kt0 + key) * (KVHEADS * HDIM) +
                 hk * HDIM + d;
      Ks[key][d] = (__bf16)kbuf[g];
      Vs[d][key] = (__bf16)vbuf[g];
    }
    __syncthreads();

    // S = Q K^T over d (two K-chunks of 32)
    v8f sfr[4] = {};
#pragma unroll
    for (int kc = 0; kc < HDIM; kc += 32) {
      v16bf a = load_frag(&Qs[0][0], LD, qr0, kc, lane);
#pragma unroll
      for (int nt = 0; nt < 4; ++nt) {
        v16bf bb = load_frag(&Ks[0][0], LD, nt * 16, kc, lane);
        sfr[nt] = wmma_bf16(a, bb, sfr[nt]);
      }
    }

    // scale + causal mask, row max / sum, online softmax update
    float mloc[8], rs[8];
#pragma unroll
    for (int v = 0; v < 8; ++v) mloc[v] = -3e38f;
#pragma unroll
    for (int nt = 0; nt < 4; ++nt) {
      float* sp = (float*)&sfr[nt];
      int colg = kt0 + nt * 16 + ln15;
#pragma unroll
      for (int v = 0; v < 8; ++v) {
        int rowg = q0 + qr0 + v + hib;
        float val = sp[v] * 8.0f + (colg > rowg ? NEGINF : 0.0f);
        sp[v] = val;
        mloc[v] = fmaxf(mloc[v], val);
      }
    }
#pragma unroll
    for (int v = 0; v < 8; ++v)
      for (int off = 1; off < 16; off <<= 1)
        mloc[v] = fmaxf(mloc[v], __shfl_xor(mloc[v], off, 32));
#pragma unroll
    for (int v = 0; v < 8; ++v) {
      float mnew = fmaxf(mr[v], mloc[v]);
      float alpha = __expf(mr[v] - mnew);
      mr[v] = mnew;
      rs[v] = 0.0f;
#pragma unroll
      for (int nt = 0; nt < 4; ++nt) {
        float* sp = (float*)&sfr[nt];
        sp[v] = __expf(sp[v] - mnew);
        rs[v] += sp[v];
      }
      for (int off = 1; off < 16; off <<= 1)
        rs[v] += __shfl_xor(rs[v], off, 32);
      lr[v] = lr[v] * alpha + rs[v];
#pragma unroll
      for (int dt = 0; dt < 4; ++dt) ((float*)&o[dt])[v] *= alpha;
    }

    // write probabilities to LDS (bf16) for the PV matmul
#pragma unroll
    for (int nt = 0; nt < 4; ++nt) {
      float* sp = (float*)&sfr[nt];
#pragma unroll
      for (int v = 0; v < 8; ++v)
        Ps[qr0 + v + hib][nt * 16 + ln15] = (__bf16)sp[v];
    }
    __syncthreads();

    // O += P V  (K dim = 64 keys, two chunks of 32)
#pragma unroll
    for (int kc = 0; kc < 64; kc += 32) {
      v16bf a = load_frag(&Ps[0][0], LD, qr0, kc, lane);
#pragma unroll
      for (int dt = 0; dt < 4; ++dt) {
        v16bf bb = load_frag(&Vs[0][0], LD, dt * 16, kc, lane);
        o[dt] = wmma_bf16(a, bb, o[dt]);
      }
    }
    __syncthreads();
  }

  // normalize and store [T][HEADS*HDIM]
#pragma unroll
  for (int dt = 0; dt < 4; ++dt) {
    float* ov = (float*)&o[dt];
#pragma unroll
    for (int v = 0; v < 8; ++v) {
      size_t row = (size_t)(b * Sq + q0 + qr0 + v + hib);
      att[row * (HEADS * HDIM) + h * HDIM + dt * 16 + ln15] =
          ov[v] / lr[v];
    }
  }
}

// ---------------------------------------------------------------------------
// MoE gate: fp32 logits, top-2, softmax weights; count tokens per expert
// ---------------------------------------------------------------------------
__global__ __launch_bounds__(256) void gate_k(const float* __restrict__ r,
                                              const float* __restrict__ gw,
                                              float* __restrict__ tw,
                                              int* __restrict__ eix,
                                              int* __restrict__ counts) {
  __shared__ float logits[NEXP];
  const size_t t = blockIdx.x;
  const int tid = threadIdx.x;
  const int e = tid >> 5, ln = tid & 31;
  float p = 0.0f;
  for (int j = ln; j < DIM; j += 32)
    p += r[t * DIM + j] * gw[(size_t)e * DIM + j];
  for (int off = 16; off > 0; off >>= 1) p += __shfl_down(p, off, 32);
  if (ln == 0) logits[e] = p;
  __syncthreads();
  if (tid == 0) {
    int i0 = 0;
    for (int e2 = 1; e2 < NEXP; ++e2)
      if (logits[e2] > logits[i0]) i0 = e2;
    int i1 = (i0 == 0) ? 1 : 0;
    for (int e2 = 0; e2 < NEXP; ++e2)
      if (e2 != i0 && logits[e2] > logits[i1]) i1 = e2;
    float l0 = logits[i0], l1 = logits[i1];
    float e1 = expf(l1 - l0);
    float w0 = 1.0f / (1.0f + e1), w1 = e1 / (1.0f + e1);
    tw[t * 2 + 0] = w0; tw[t * 2 + 1] = w1;
    eix[t * 2 + 0] = i0; eix[t * 2 + 1] = i1;
    atomicAdd(&counts[i0], 1);
    atomicAdd(&counts[i1], 1);
  }
}

__global__ void init_k(int* counts, int* basep, int* cursor) {
  int i = threadIdx.x;
  if (i < NEXP) { counts[i] = 0; basep[i] = 0; cursor[i] = 0; }
}

__global__ void scan_k(const int* counts, int* basep, int* cursor) {
  if (threadIdx.x == 0) {
    int acc = 0;
    for (int e = 0; e < NEXP; ++e) {
      basep[e] = acc; cursor[e] = acc; acc += counts[e];
    }
  }
}

__global__ __launch_bounds__(256) void fill_k(const int* __restrict__ eix,
                                              const float* __restrict__ tw,
                                              int* __restrict__ cursor,
                                              int* __restrict__ slotTok,
                                              float* __restrict__ sw,
                                              int* __restrict__ slots) {
  int t = blockIdx.x * 256 + threadIdx.x;
  if (t >= Ttok) return;
#pragma unroll
  for (int k = 0; k < TOPK; ++k) {
    int e = eix[t * 2 + k];
    int pos = atomicAdd(&cursor[e], 1);
    slotTok[pos] = t;
    sw[pos] = tw[t * 2 + k];
    slots[t * 2 + k] = pos;
  }
}

__global__ __launch_bounds__(256) void combine_k(const float* __restrict__ h,
                                                 const float* __restrict__ y,
                                                 const int* __restrict__ slots,
                                                 float* __restrict__ out) {
  size_t idx = (size_t)blockIdx.x * 256 + threadIdx.x;
  if (idx >= (size_t)Ttok * DIM) return;
  size_t t = idx >> 10;
  int d = idx & (DIM - 1);
  out[idx] = h[idx] + y[(size_t)slots[t * 2 + 0] * DIM + d] +
             y[(size_t)slots[t * 2 + 1] * DIM + d];
}

// ---------------------------------------------------------------------------
// Host side
// ---------------------------------------------------------------------------
extern "C" void kernel_launch(void* const* d_in, const int* in_sizes, int n_in,
                              void* d_out, int out_size, void* d_ws,
                              size_t ws_size, hipStream_t stream) {
  const float* x       = (const float*)d_in[0];
  const float* fcos    = (const float*)d_in[1];
  const float* fsin    = (const float*)d_in[2];
  const float* attn_nw = (const float*)d_in[6];
  const float* ffn_nw  = (const float*)d_in[7];
  const float* wq      = (const float*)d_in[8];
  const float* wk      = (const float*)d_in[9];
  const float* wv      = (const float*)d_in[10];
  const float* wo      = (const float*)d_in[11];
  const float* gate_w  = (const float*)d_in[12];
  const float* w1      = (const float*)d_in[13];
  const float* w2      = (const float*)d_in[14];
  const float* w3      = (const float*)d_in[15];

  // workspace layout (floats)
  float* wf = (float*)d_ws;
  size_t off = 0;
  float* xn_r = wf + off; off += (size_t)Ttok * DIM;           // xn, later r
  float* qb   = wf + off; off += (size_t)Ttok * HEADS * HDIM;
  float* kb   = wf + off; off += (size_t)Ttok * KVHEADS * HDIM;
  float* vb   = wf + off; off += (size_t)Ttok * KVHEADS * HDIM;
  float* att  = wf + off; off += (size_t)Ttok * HEADS * HDIM;
  float* hb   = wf + off; off += (size_t)Ttok * DIM;
  float* h1   = wf + off; off += (size_t)NSLOT * HIDDEN;
  float* yb   = wf + off; off += (size_t)NSLOT * DIM;
  float* tw   = wf + off; off += (size_t)Ttok * TOPK;
  float* sw   = wf + off; off += (size_t)NSLOT;
  int* ip     = (int*)(wf + off);
  int* eix     = ip;                 ip += Ttok * TOPK;
  int* slots   = ip;                 ip += Ttok * TOPK;
  int* slotTok = ip;                 ip += NSLOT;
  int* counts  = ip;                 ip += NEXP;
  int* basep   = ip;                 ip += NEXP;
  int* cursor  = ip;                 ip += NEXP;

  float* outp = (float*)d_out;

  // 0) zero routing state
  hipLaunchKernelGGL(init_k, dim3(1), dim3(32), 0, stream, counts, basep,
                     cursor);

  // 1) attn rmsnorm
  hipLaunchKernelGGL(rmsnorm_k, dim3(Ttok), dim3(256), 0, stream, x, attn_nw,
                     xn_r);

  // 2) Q/K/V projections (C = xn @ W^T)
  hipLaunchKernelGGL((gemm_wmma_k<0, false>), dim3(8, 64, 1), dim3(256), 0,
                     stream, xn_r, wq, qb, Ttok, HEADS * HDIM, DIM, (size_t)0,
                     nullptr, nullptr, nullptr, nullptr, nullptr);
  hipLaunchKernelGGL((gemm_wmma_k<0, false>), dim3(2, 64, 1), dim3(256), 0,
                     stream, xn_r, wk, kb, Ttok, KVHEADS * HDIM, DIM, (size_t)0,
                     nullptr, nullptr, nullptr, nullptr, nullptr);
  hipLaunchKernelGGL((gemm_wmma_k<0, false>), dim3(2, 64, 1), dim3(256), 0,
                     stream, xn_r, wv, vb, Ttok, KVHEADS * HDIM, DIM, (size_t)0,
                     nullptr, nullptr, nullptr, nullptr, nullptr);

  // 3) RoPE on q and k (in place)
  hipLaunchKernelGGL(rope_k, dim3((Ttok * HEADS * 32 + 255) / 256), dim3(256),
                     0, stream, qb, HEADS, fcos, fsin);
  hipLaunchKernelGGL(rope_k, dim3((Ttok * KVHEADS * 32 + 255) / 256),
                     dim3(256), 0, stream, kb, KVHEADS, fcos, fsin);

  // 4) flash attention
  hipLaunchKernelGGL(attn_k, dim3(Sq / 64, HEADS, Bsz), dim3(128), 0, stream,
                     qb, kb, vb, att);

  // 5) output projection + residual: h = x + att @ wo^T
  hipLaunchKernelGGL((gemm_wmma_k<1, false>), dim3(8, 64, 1), dim3(256), 0,
                     stream, att, wo, hb, Ttok, DIM, HEADS * HDIM, (size_t)0,
                     nullptr, nullptr, nullptr, x, nullptr);

  // 6) ffn rmsnorm -> r (reuse xn buffer)
  hipLaunchKernelGGL(rmsnorm_k, dim3(Ttok), dim3(256), 0, stream, hb, ffn_nw,
                     xn_r);

  // 7) gate + routing
  hipLaunchKernelGGL(gate_k, dim3(Ttok), dim3(256), 0, stream, xn_r, gate_w,
                     tw, eix, counts);
  hipLaunchKernelGGL(scan_k, dim3(1), dim3(32), 0, stream, counts, basep,
                     cursor);
  hipLaunchKernelGGL(fill_k, dim3((Ttok + 255) / 256), dim3(256), 0, stream,
                     eix, tw, cursor, slotTok, sw, slots);

  // 8) experts (routed): h1 = r@w3^T ; h1 = silu(r@w1^T)*h1 ; y = sw*(h1@w2)
  const size_t wstride = (size_t)HIDDEN * DIM;
  hipLaunchKernelGGL((gemm_wmma_k<0, false>), dim3(HIDDEN / 128, 64, NEXP),
                     dim3(256), 0, stream, xn_r, w3, h1, Ttok, HIDDEN, DIM,
                     wstride, basep, counts, slotTok, nullptr, nullptr);
  hipLaunchKernelGGL((gemm_wmma_k<2, false>), dim3(HIDDEN / 128, 64, NEXP),
                     dim3(256), 0, stream, xn_r, w1, h1, Ttok, HIDDEN, DIM,
                     wstride, basep, counts, slotTok, nullptr, nullptr);
  hipLaunchKernelGGL((gemm_wmma_k<3, true>), dim3(DIM / 128, 64, NEXP),
                     dim3(256), 0, stream, h1, w2, yb, Ttok, DIM, HIDDEN,
                     wstride, basep, counts, nullptr, nullptr, sw);

  // 9) combine: out = h + sum_k w_k * y_k
  hipLaunchKernelGGL(combine_k, dim3(((size_t)Ttok * DIM + 255) / 256),
                     dim3(256), 0, stream, hb, yb, slots, outp);
}
